// RawWaveformMamba_67705864454290
// MI455X (gfx1250) — matrix operand order repaired and checked by reading, not compiled, not run on hardware
//
#include <hip/hip_runtime.h>
#include <hip/hip_bf16.h>
#include <utility>

// ---------------- model constants ----------------
#define NLAYER 6
#define DMODEL 256
#define DSTATE 32
#define DCONV  4
#define DINNER 512
#define DTRANK 16
#define DBLC   (DTRANK + 2 * DSTATE)   // 80
#define BATCH  8
#define SEQL   1024
#define MROWS  (BATCH * SEQL)          // 8192
#define NCLS   50
#define PATCH  160

typedef __attribute__((ext_vector_type(16))) _Float16 v16h;
typedef __attribute__((ext_vector_type(8)))  _Float16 v8h;
typedef __attribute__((ext_vector_type(8)))  float    v8f;
typedef __attribute__((ext_vector_type(4)))  float    v4f;

__device__ __forceinline__ float siluf(float v) {
    return v / (1.0f + __expf(-v));
}
__device__ __forceinline__ float softplusf(float v) {
    return (v > 20.0f) ? v : log1pf(__expf(v));
}

// ---------------------------------------------------------------------------
// Weight prep: W (N,K) f32 row-major  ->  Wt (Kpad,N) f16, zero-padded rows
// for k >= K. Makes GEMM B-fragment loads coalesced across lanes (n contiguous) and removes all
// K-guards from the hot loop (padded weights annihilate garbage A columns).
// ---------------------------------------------------------------------------
__global__ void prep_weight(const float* __restrict__ W,
                            _Float16* __restrict__ Wt,
                            int N, int K, int total /* Kpad*N */) {
    const int idx = blockIdx.x * blockDim.x + threadIdx.x;
    if (idx >= total) return;
    const int n = idx % N;
    const int k = idx / N;
    Wt[idx] = (k < K) ? (_Float16)W[(size_t)n * K + k] : (_Float16)0.0f;
}

// ---------------------------------------------------------------------------
// WMMA GEMM:  C[m, coff+n] = epilogue( sum_k A[m,k] * Wt[k,n] )
//   A: f32 row-major (lda, 64B-aligned rows); Wt: f16 K-major (Kpad x N).
//   One wave per MT x NT block of 16x16 tiles; grid = (N/(16*NT), M/(16*MT)).
//   Branch-free inner loop: 4*MT float4 A-loads + 16*NT scalar f16 B-loads + MT*NT
//   WMMAs per 32-wide K step. Kpad must be a multiple of 32.
//   act: 0 = none, 1 = softplus. bias/res nullable (uniform branches only).
// ---------------------------------------------------------------------------
template <int MT, int NT>
__global__ void wmma_gemm(const float* __restrict__ A, int lda,
                          const _Float16* __restrict__ Wt, int N,
                          const float* __restrict__ bias,
                          const float* __restrict__ res, int ldres,
                          float* __restrict__ C, int ldc, int coff,
                          int Kpad, int act) {
    const int lane = threadIdx.x;            // 0..31, full wave active
    const int tn0  = blockIdx.x * NT;
    const int tm0  = blockIdx.y * MT;

    // A fragment (16-bit A, 16x32): lanes 0-15 rows M=lane&15, K=0..15;
    // lanes 16-31 same rows, K=16..31.
    const int amr = lane & 15;
    const int akb = (lane >> 4) * 16;
    // B fragment: K = 16*(lane>>4) + element, N = lane & 15.
    // C/D: VGPR v -> M = v + 8*(lane>>4); N = lane & 15.

    v8f zero = {};
    v8f acc[MT][NT];
    #pragma unroll
    for (int i = 0; i < MT; ++i)
        #pragma unroll
        for (int j = 0; j < NT; ++j) acc[i][j] = zero;

    for (int k0 = 0; k0 < Kpad; k0 += 32) {
        v16h af[MT];
        #pragma unroll
        for (int i = 0; i < MT; ++i) {
            const float* p = A + (size_t)((tm0 + i) * 16 + amr) * lda + k0 + akb;
            if (k0 + 32 < Kpad) __builtin_prefetch(p + 32, 0, 3);
            const v4f a0 = *(const v4f*)(p);
            const v4f a1 = *(const v4f*)(p + 4);
            const v4f a2 = *(const v4f*)(p + 8);
            const v4f a3 = *(const v4f*)(p + 12);
            #pragma unroll
            for (int e = 0; e < 4; ++e) {
                af[i][e]      = (_Float16)a0[e];
                af[i][4 + e]  = (_Float16)a1[e];
                af[i][8 + e]  = (_Float16)a2[e];
                af[i][12 + e] = (_Float16)a3[e];
            }
        }
        v16h bf[NT];
        const _Float16* wrow = Wt + (size_t)(k0 + akb) * N + amr;
        #pragma unroll
        for (int j = 0; j < NT; ++j) {
            #pragma unroll
            for (int e = 0; e < 16; ++e)
                bf[j][e] = wrow[(size_t)e * N + (tn0 + j) * 16];
        }
        #pragma unroll
        for (int i = 0; i < MT; ++i)
            #pragma unroll
            for (int j = 0; j < NT; ++j)
                acc[i][j] = __builtin_amdgcn_wmma_f32_16x16x32_f16(
                    false, af[i], false, bf[j], (short)0, acc[i][j],
                    false, false);
    }

    const int rhi = (lane >> 4) * 8;
    #pragma unroll
    for (int i = 0; i < MT; ++i) {
        #pragma unroll
        for (int j = 0; j < NT; ++j) {
            const int n  = (tn0 + j) * 16 + (lane & 15);
            const float bv = bias ? bias[n] : 0.0f;
            #pragma unroll
            for (int v = 0; v < 8; ++v) {
                const int m = (tm0 + i) * 16 + rhi + v;
                float val = acc[i][j][v] + bv;
                if (act == 1) val = softplusf(val);
                if (res) val += res[(size_t)m * ldres + n];
                C[(size_t)m * ldc + coff + n] = val;
            }
        }
    }
}

// ---------------------------------------------------------------------------
// Row LayerNorm over 256 features. One wave per row (wave32: 8 elems/lane).
// ---------------------------------------------------------------------------
__global__ void layernorm_rows(const float* __restrict__ X,
                               const float* __restrict__ g,
                               const float* __restrict__ b,
                               float* __restrict__ Y, int nrows) {
    const int wave = threadIdx.x >> 5;
    const int lane = threadIdx.x & 31;
    const int row  = blockIdx.x * (blockDim.x >> 5) + wave;
    if (row >= nrows) return;
    const float* x = X + (size_t)row * DMODEL;

    float v[8];
    float s = 0.0f;
    #pragma unroll
    for (int i = 0; i < 8; ++i) { v[i] = x[lane + 32 * i]; s += v[i]; }
    #pragma unroll
    for (int m = 16; m; m >>= 1) s += __shfl_xor(s, m, 32);
    const float mu = s * (1.0f / DMODEL);

    float q = 0.0f;
    #pragma unroll
    for (int i = 0; i < 8; ++i) { const float d = v[i] - mu; q += d * d; }
    #pragma unroll
    for (int m = 16; m; m >>= 1) q += __shfl_xor(q, m, 32);
    const float inv = rsqrtf(q * (1.0f / DMODEL) + 1e-5f);

    float* y = Y + (size_t)row * DMODEL;
    #pragma unroll
    for (int i = 0; i < 8; ++i) {
        const int c = lane + 32 * i;
        y[c] = (v[i] - mu) * inv * g[c] + b[c];
    }
}

// ---------------------------------------------------------------------------
// Depthwise causal conv (k=4) + bias + SiLU. rev=1 => anti-causal w/ flipped
// taps (== reference conv applied to the reversed sequence, indexed back).
// x_p lives in cols [0,512) of the (M,1024) xz buffer.
// ---------------------------------------------------------------------------
__global__ void dwconv_silu(const float* __restrict__ xz,
                            const float* __restrict__ cw,  // (512,4)
                            const float* __restrict__ cb,  // (512)
                            float* __restrict__ xc,        // (M,512)
                            int rev) {
    const size_t idx = (size_t)blockIdx.x * blockDim.x + threadIdx.x;
    const int d = (int)(idx & (DINNER - 1));
    const int t = (int)((idx >> 9) & (SEQL - 1));
    const int b = (int)(idx >> 19);

    float acc = cb[d];
    #pragma unroll
    for (int j = 0; j < DCONV; ++j) {
        const int tt = rev ? (t + (DCONV - 1) - j) : (t - (DCONV - 1) + j);
        if (tt >= 0 && tt < SEQL)
            acc += cw[d * DCONV + j] *
                   xz[((size_t)b * SEQL + tt) * (2 * DINNER) + d];
    }
    xc[idx] = siluf(acc);
}

// ---------------------------------------------------------------------------
// Selective scan. 16 blocks (b, half) x 256 threads; each thread owns one
// channel d: 32-wide state + precomputed A row in registers, marches over
// time (forward or reverse). B_t / C_t staged via LDS once per step.
// Fused epilogue: yg = (y + xc*D) * silu(z).
// ---------------------------------------------------------------------------
__global__ void mamba_scan(const float* __restrict__ dt,    // (M,512)
                           const float* __restrict__ dbl,   // (M,80)
                           const float* __restrict__ xc,    // (M,512)
                           const float* __restrict__ xz,    // (M,1024), z @+512
                           const float* __restrict__ A_log, // (512,32)
                           const float* __restrict__ Dp,    // (512)
                           float* __restrict__ yg,          // (M,512)
                           int rev) {
    __shared__ float sB[DSTATE];
    __shared__ float sC[DSTATE];
    const int b = blockIdx.x >> 1;
    const int d = ((blockIdx.x & 1) << 8) + threadIdx.x;

    float Av[DSTATE], hst[DSTATE];
    #pragma unroll
    for (int s = 0; s < DSTATE; ++s) {
        Av[s]  = -__expf(A_log[d * DSTATE + s]);
        hst[s] = 0.0f;
    }
    const float dp = Dp[d];

    for (int step = 0; step < SEQL; ++step) {
        const int t = rev ? (SEQL - 1 - step) : step;
        const size_t row = (size_t)b * SEQL + t;
        __syncthreads();
        if (threadIdx.x < 2 * DSTATE) {
            const float val = dbl[row * DBLC + DTRANK + threadIdx.x];
            if (threadIdx.x < DSTATE) sB[threadIdx.x] = val;
            else                      sC[threadIdx.x - DSTATE] = val;
        }
        __syncthreads();

        const float dtv = dt[row * DINNER + d];
        const float xv  = xc[row * DINNER + d];
        const float dbx = dtv * xv;
        float y = 0.0f;
        #pragma unroll
        for (int s = 0; s < DSTATE; ++s) {
            const float dA = __expf(dtv * Av[s]);
            hst[s] = dA * hst[s] + dbx * sB[s];
            y += hst[s] * sC[s];
        }
        const float zv = xz[row * (2 * DINNER) + DINNER + d];
        yg[row * DINNER + d] = (y + xv * dp) * siluf(zv);
    }
}

// ---------------------------------------------------------------------------
// Mean pool over L and tiny classifier.
// ---------------------------------------------------------------------------
__global__ void pool_mean(const float* __restrict__ X, float* __restrict__ out) {
    const int idx = blockIdx.x * blockDim.x + threadIdx.x;  // 0..2047
    const int b = idx >> 8, c = idx & (DMODEL - 1);
    float s = 0.0f;
    for (int t = 0; t < SEQL; ++t)
        s += X[((size_t)b * SEQL + t) * DMODEL + c];
    out[idx] = s * (1.0f / SEQL);
}

__global__ void classifier(const float* __restrict__ pooled,
                           const float* __restrict__ Wc,
                           const float* __restrict__ bc,
                           float* __restrict__ out) {
    const int idx = blockIdx.x * blockDim.x + threadIdx.x;
    if (idx >= BATCH * NCLS) return;
    const int b = idx / NCLS, c = idx % NCLS;
    float s = bc[c];
    for (int d = 0; d < DMODEL; ++d)
        s += pooled[b * DMODEL + d] * Wc[c * DMODEL + d];
    out[idx] = s;
}

// ---------------------------------------------------------------------------
extern "C" void kernel_launch(void* const* d_in, const int* in_sizes, int n_in,
                              void* d_out, int out_size, void* d_ws, size_t ws_size,
                              hipStream_t stream) {
    (void)in_sizes; (void)n_in; (void)out_size; (void)ws_size;

    const float* x     = (const float*)d_in[0];
    const float* pe_w  = (const float*)d_in[1];
    const float* pe_b  = (const float*)d_in[2];
    const float* in_g  = (const float*)d_in[3];
    const float* in_b  = (const float*)d_in[4];
    const float* ng    = (const float*)d_in[5];
    const float* nb    = (const float*)d_in[6];
    const float* Win   = (const float*)d_in[7];
    const float* cw    = (const float*)d_in[8];
    const float* cb    = (const float*)d_in[9];
    const float* Wx    = (const float*)d_in[10];
    const float* Wdt   = (const float*)d_in[11];
    const float* bdt   = (const float*)d_in[12];
    const float* A_log = (const float*)d_in[13];
    const float* Dp    = (const float*)d_in[14];
    const float* Wout  = (const float*)d_in[15];
    const float* Wblk  = (const float*)d_in[16];
    const float* bblk  = (const float*)d_in[17];
    const float* fg    = (const float*)d_in[18];
    const float* fb    = (const float*)d_in[19];
    const float* Wc    = (const float*)d_in[20];
    const float* bc    = (const float*)d_in[21];
    float* out = (float*)d_out;

    // -------- workspace layout (f32 activations) --------
    float* ws   = (float*)d_ws;
    float* h    = ws;                                     // (M,256)
    float* hn   = h    + (size_t)MROWS * DMODEL;          // (M,256)
    float* xzb  = hn   + (size_t)MROWS * DMODEL;          // (M,1024)
    float* xcb  = xzb  + (size_t)MROWS * (2 * DINNER);    // (M,512)
    float* dblb = xcb  + (size_t)MROWS * DINNER;          // (M,80)
    float* dtb  = dblb + (size_t)MROWS * DBLC;            // (M,512)
    float* ygb  = dtb  + (size_t)MROWS * DINNER;          // (M,512)
    float* comb = ygb  + (size_t)MROWS * DINNER;          // (M,512)
    float* pool = comb + (size_t)MROWS * DINNER;          // (8,256)

    // -------- f16 K-major weight region --------
    _Float16* f16base = (_Float16*)(pool + 2048);
    _Float16* pe_t    = f16base;                                 // 160x256
    _Float16* Win_t   = pe_t  + (size_t)PATCH * DMODEL;          // 12 x 256x1024
    _Float16* Wx_t    = Win_t + (size_t)12 * DMODEL * (2*DINNER);// 12 x 512x80
    _Float16* Wdt_t   = Wx_t  + (size_t)12 * DINNER * DBLC;      // 12 x 32x512
    _Float16* Wout_t  = Wdt_t + (size_t)12 * 32 * DINNER;        // 12 x 512x256
    _Float16* Wblk_t  = Wout_t+ (size_t)12 * DINNER * DMODEL;    // 6 x 512x256

    auto prep = [&](const float* W, _Float16* Wt, int N, int K, int Kpad) {
        const int total = Kpad * N;
        prep_weight<<<(total + 255) / 256, 256, 0, stream>>>(W, Wt, N, K, total);
    };

    // One-time (per launch) weight transpose + f16 convert, K padded to 32.
    prep(pe_w, pe_t, DMODEL, PATCH, PATCH);
    for (int lr = 0; lr < 12; ++lr) {
        prep(Win  + (size_t)lr * (2*DINNER) * DMODEL,
             Win_t  + (size_t)lr * DMODEL * (2*DINNER), 2*DINNER, DMODEL, DMODEL);
        prep(Wx   + (size_t)lr * DBLC * DINNER,
             Wx_t   + (size_t)lr * DINNER * DBLC, DBLC, DINNER, DINNER);
        prep(Wdt  + (size_t)lr * DINNER * DTRANK,
             Wdt_t  + (size_t)lr * 32 * DINNER, DINNER, DTRANK, 32);
        prep(Wout + (size_t)lr * DMODEL * DINNER,
             Wout_t + (size_t)lr * DINNER * DMODEL, DMODEL, DINNER, DINNER);
    }
    for (int i = 0; i < NLAYER; ++i)
        prep(Wblk + (size_t)i * DMODEL * (2*DMODEL),
             Wblk_t + (size_t)i * (2*DMODEL) * DMODEL, DMODEL, 2*DMODEL, 2*DMODEL);

    auto gemm22 = [&](const float* A, int lda, const _Float16* Wt,
                      const float* bias, const float* res, int ldres,
                      float* C, int ldc, int coff, int N, int Kpad, int act) {
        dim3 grid(N / 32, MROWS / 32);
        wmma_gemm<2, 2><<<grid, 32, 0, stream>>>(A, lda, Wt, N, bias, res,
                                                 ldres, C, ldc, coff, Kpad, act);
    };
    auto gemm41 = [&](const float* A, int lda, const _Float16* Wt,
                      const float* bias, const float* res, int ldres,
                      float* C, int ldc, int coff, int N, int Kpad, int act) {
        dim3 grid(N / 16, MROWS / 64);
        wmma_gemm<4, 1><<<grid, 32, 0, stream>>>(A, lda, Wt, N, bias, res,
                                                 ldres, C, ldc, coff, Kpad, act);
    };

    // 1) patch embed: x viewed as (8192,160) patches @ pe_t -> h (M,256)
    gemm22(x, PATCH, pe_t, pe_b, nullptr, 0, h, DMODEL, 0, DMODEL, PATCH, 0);
    // 2) input layernorm (in place)
    layernorm_rows<<<MROWS / 8, 256, 0, stream>>>(h, in_g, in_b, h, MROWS);

    for (int i = 0; i < NLAYER; ++i) {
        layernorm_rows<<<MROWS / 8, 256, 0, stream>>>(h, ng + i * DMODEL,
                                                      nb + i * DMODEL, hn, MROWS);
        for (int r = 0; r < 2; ++r) {
            const size_t lr = (size_t)(i * 2 + r);
            const _Float16* Win_p  = Win_t  + lr * DMODEL * (2*DINNER);
            const _Float16* Wx_p   = Wx_t   + lr * DINNER * DBLC;
            const _Float16* Wdt_p  = Wdt_t  + lr * 32 * DINNER;
            const _Float16* Wout_p = Wout_t + lr * DINNER * DMODEL;
            const float* cw_p   = cw    + lr * DINNER * DCONV;
            const float* cb_p   = cb    + lr * DINNER;
            const float* bdt_p  = bdt   + lr * DINNER;
            const float* Alog_p = A_log + lr * DINNER * DSTATE;
            const float* Dp_p   = Dp    + lr * DINNER;

            // in_proj: (M,256) @ Wt(256,1024) -> xz (M,1024)
            gemm22(hn, DMODEL, Win_p, nullptr, nullptr, 0,
                   xzb, 2 * DINNER, 0, 2 * DINNER, DMODEL, 0);
            // depthwise conv + bias + SiLU (direction-aware)
            dwconv_silu<<<(MROWS * DINNER) / 256, 256, 0, stream>>>(
                xzb, cw_p, cb_p, xcb, r);
            // x_proj: (M,512) @ Wt(512,80) -> dbl (M,80); N=80 -> 4x1 tiles
            gemm41(xcb, DINNER, Wx_p, nullptr, nullptr, 0,
                   dblb, DBLC, 0, DBLC, DINNER, 0);
            // dt_proj + softplus: dbl[:, 0:16] @ Wt(32pad,512) + bdt -> dt
            gemm22(dblb, DBLC, Wdt_p, bdt_p, nullptr, 0,
                   dtb, DINNER, 0, DINNER, 32, 1);
            // selective scan + gating (direction-aware)
            mamba_scan<<<BATCH * 2, 256, 0, stream>>>(
                dtb, dblb, xcb, xzb, Alog_p, Dp_p, ygb, r);
            // out_proj: (M,512) @ Wt(512,256) -> comb[:, r*256 :]
            gemm22(ygb, DINNER, Wout_p, nullptr, nullptr, 0,
                   comb, 2 * DMODEL, r * DMODEL, DMODEL, DINNER, 0);
        }
        // block mixer + bias + residual: comb (M,512) @ Wt(512,256) + h -> hn
        gemm22(comb, 2 * DMODEL, Wblk_t + (size_t)i * (2*DMODEL) * DMODEL,
               bblk + i * DMODEL, h, DMODEL,
               hn, DMODEL, 0, DMODEL, 2 * DMODEL, 0);
        std::swap(h, hn);
    }

    // final norm, mean pool over L, classifier
    layernorm_rows<<<MROWS / 8, 256, 0, stream>>>(h, fg, fb, hn, MROWS);
    pool_mean<<<BATCH, 256, 0, stream>>>(hn, pool);
    classifier<<<2, 256, 0, stream>>>(pool, Wc, bc, out);
}
